// NIC_language_model_22333829939697
// MI455X (gfx1250) — compile-verified, matrix operand order checked by reading, not run on hardware
//
#include <hip/hip_runtime.h>
#include <hip/hip_bf16.h>

// ---------------- types for CDNA5 WMMA ----------------
typedef __attribute__((ext_vector_type(16))) __bf16 v16bf;
typedef __attribute__((ext_vector_type(8)))  __bf16 v8bf;
typedef __attribute__((ext_vector_type(8)))  float  v8f;

// ---------------- problem constants ----------------
#define B_   128
#define T_   24
#define S_   23       // scan length = T-1
#define E_   512
#define R_   2048
#define V_   32000
#define G4E  2048     // 4*E
#define KZ   1024     // K of [x|h] concat GEMM

// ---------------- workspace layout (bytes) ----------------
static const size_t OFF_SORT  = 0;                                   // 128 * int
static const size_t OFF_IMGBF = 1024;                                // 128*2048 bf16 = 512KB
static const size_t OFF_WIMG  = OFF_IMGBF + (size_t)B_*R_*2;         // 512*2048 bf16 = 2MB
static const size_t OFF_WZ    = OFF_WIMG  + (size_t)E_*R_*2;         // 2048*1024 bf16 = 4MB
static const size_t OFF_XBF   = OFF_WZ    + (size_t)G4E*KZ*2;        // 24*128*512 bf16 = 3MB
static const size_t OFF_HS    = OFF_XBF   + (size_t)T_*B_*E_*2;      // 23*128*512 f32 = 6MB
static const size_t OFF_HSBF  = OFF_HS    + (size_t)S_*B_*E_*4;      // 23*128*512 bf16 = 3MB
static const size_t OFF_WDEC  = OFF_HSBF  + (size_t)S_*B_*E_*2;      // 32000*512 bf16 = 32.8MB

// output layout (elements)
static const size_t OUT_CAP  = (size_t)B_*S_*V_;   // preds first: 94,208,000
static const size_t OUT_DEC  = OUT_CAP + (size_t)B_*T_;
static const size_t OUT_SORT = OUT_DEC + B_;

// ---------------- helpers ----------------
__device__ __forceinline__ v8f zero8() {
  v8f v;
#pragma unroll
  for (int i = 0; i < 8; ++i) v[i] = 0.f;
  return v;
}

// Load one 16x32 bf16 A/B fragment from a row-major matrix.
// lane<16  : row rbase+lane,    k = kbase+[0..7], kbase+[16..23]
// lane>=16 : row rbase+lane-16, k = kbase+[8..15], kbase+[24..31]
__device__ __forceinline__ v16bf load_frag(const __bf16* base, int stride,
                                           int rbase, int kbase, int lane) {
  int r  = rbase + (lane & 15);
  int k0 = kbase + ((lane & 16) ? 8 : 0);
  const __bf16* p = base + (size_t)r * stride + k0;
  v8bf lo = *(const v8bf*)(p);
  v8bf hi = *(const v8bf*)(p + 16);
  return __builtin_shufflevector(lo, hi, 0,1,2,3,4,5,6,7,8,9,10,11,12,13,14,15);
}

__device__ __forceinline__ v8f wmma_bf16(v16bf a, v16bf b, v8f c) {
  return __builtin_amdgcn_wmma_f32_16x16x32_bf16(false, a, false, b,
                                                 (short)0, c, false, false);
}

__device__ __forceinline__ float sigmoidf_(float x) {
  return 1.0f / (1.0f + __expf(-x));
}

// ---------------- kernel 1: sort + int outputs ----------------
__global__ void nic_prep(const int* __restrict__ cl, const int* __restrict__ cap,
                         int* __restrict__ sort_ind_g, int* __restrict__ outi) {
  __shared__ int s_cl[B_];
  __shared__ int s_ind[B_];
  int i = threadIdx.x;
  s_cl[i] = cl[i];
  __syncthreads();
  int my = s_cl[i];
  int rank = 0;
  for (int j = 0; j < B_; ++j) {
    int cj = s_cl[j];
    if (cj > my || (cj == my && j < i)) rank++;
  }
  s_ind[rank] = i;     // stable argsort of -cl
  __syncthreads();
  int src = s_ind[i];
  sort_ind_g[i] = src;
  outi[OUT_SORT + i] = src;
  outi[OUT_DEC + i]  = s_cl[src] - 1;
  for (int t = 0; t < T_; ++t)
    outi[OUT_CAP + (size_t)i * T_ + t] = cap[src * T_ + t];
}

// ---------------- kernel 2: weight conversion to bf16 ----------------
__global__ void nic_convert(const float* __restrict__ W_img,
                            const float* __restrict__ W_ih,
                            const float* __restrict__ W_hh,
                            const float* __restrict__ W_dec,
                            __bf16* __restrict__ Wimg_bf,
                            __bf16* __restrict__ Wz_bf,
                            __bf16* __restrict__ Wdec_bf) {
  const size_t N1 = (size_t)E_ * R_;
  const size_t N2 = (size_t)G4E * KZ;
  const size_t N3 = (size_t)V_ * E_;
  const size_t total = N1 + N2 + N3;
  for (size_t i = (size_t)blockIdx.x * blockDim.x + threadIdx.x; i < total;
       i += (size_t)gridDim.x * blockDim.x) {
    if (i < N1) {
      Wimg_bf[i] = (__bf16)W_img[i];
    } else if (i < N1 + N2) {
      size_t i2 = i - N1;
      int n = (int)(i2 >> 10), k = (int)(i2 & 1023);
      float v = (k < E_) ? W_ih[(size_t)n * E_ + k] : W_hh[(size_t)n * E_ + (k - E_)];
      Wz_bf[i2] = (__bf16)v;
    } else {
      size_t i3 = i - N1 - N2;
      Wdec_bf[i3] = (__bf16)W_dec[i3];
    }
  }
}

// ---------------- kernel 3: gather img (sorted) + embeddings ----------------
__global__ void nic_gather(const float* __restrict__ img,
                           const int* __restrict__ cap,
                           const float* __restrict__ emb,
                           const int* __restrict__ sort_ind,
                           __bf16* __restrict__ img_bf,
                           __bf16* __restrict__ x_bf) {
  const size_t N1 = (size_t)B_ * R_;               // sorted image rows
  const size_t N2 = (size_t)S_ * B_ * E_;          // embeddings for steps 1..23
  const size_t total = N1 + N2;
  for (size_t i = (size_t)blockIdx.x * blockDim.x + threadIdx.x; i < total;
       i += (size_t)gridDim.x * blockDim.x) {
    if (i < N1) {
      int b = (int)(i >> 11), k = (int)(i & 2047);
      img_bf[i] = (__bf16)img[(size_t)sort_ind[b] * R_ + k];
    } else {
      size_t i2 = i - N1;
      int tp = (int)(i2 / ((size_t)B_ * E_));      // 0..22 -> step t = tp+1
      int rem = (int)(i2 - (size_t)tp * B_ * E_);
      int b = rem >> 9, e = rem & 511;
      int tok = cap[sort_ind[b] * T_ + tp];
      x_bf[((size_t)(tp + 1) * B_ + b) * E_ + e] = (__bf16)emb[(size_t)tok * E_ + e];
    }
  }
}

// ---------------- kernel 4: x0 = img_s @ W_img.T + b_img  (WMMA) ----------------
__global__ void __launch_bounds__(256)
nic_x0_gemm(const __bf16* __restrict__ img_bf, const __bf16* __restrict__ Wimg_bf,
            const float* __restrict__ b_img, __bf16* __restrict__ x0_bf) {
  int tid = threadIdx.x, wave = tid >> 5, lane = tid & 31;
  int wid = blockIdx.x * 8 + wave;           // 256 tiles: 8 M-tiles x 32 N-tiles
  int mt = wid >> 5, nt = wid & 31;
  int mbase = mt * 16, nbase = nt * 16;
  v8f acc = zero8();
  for (int kb = 0; kb < R_; kb += 32) {
    v16bf a = load_frag(img_bf,  R_, mbase, kb, lane);
    v16bf b = load_frag(Wimg_bf, R_, nbase, kb, lane);
    acc = wmma_bf16(a, b, acc);
  }
  int n = nbase + (lane & 15);
  int m0 = mbase + ((lane >> 4) << 3);
  float bias = b_img[n];
#pragma unroll
  for (int i = 0; i < 8; ++i)
    x0_bf[(size_t)(m0 + i) * E_ + n] = (__bf16)(acc[i] + bias);
}

// ---------------- kernel 5: persistent LSTM scan (8 blocks, 24 steps) ----------------
__global__ void __launch_bounds__(256)
nic_lstm(const __bf16* __restrict__ x_bf, const __bf16* __restrict__ Wz,
         const float* __restrict__ b_ih, const float* __restrict__ b_hh,
         float* __restrict__ hs) {
  __shared__ __bf16 zsm[16 * KZ];                  // [row][k]: k<512 = x, k>=512 = h
  int tid = threadIdx.x, wave = tid >> 5, lane = tid & 31;
  int mrow = blockIdx.x * 16;                      // global batch-row base

  v8f cst[4];                                      // c state, one 16x16 tile per e-tile
  v8f hreg[4];
#pragma unroll
  for (int j = 0; j < 4; ++j) cst[j] = zero8();

  for (int idx = tid; idx < 16 * E_; idx += 256) { // h0 = 0
    int r = idx >> 9, e = idx & 511;
    zsm[r * KZ + E_ + e] = (__bf16)0.f;
  }

  for (int t = 0; t < T_; ++t) {
    __syncthreads();
    const __bf16* xsrc = x_bf + ((size_t)t * B_ + mrow) * E_;
    for (int idx = tid; idx < 16 * E_; idx += 256) {
      int r = idx >> 9, e = idx & 511;
      zsm[r * KZ + e] = xsrc[(size_t)r * E_ + e];
    }
    __syncthreads();

    for (int j = 0; j < 4; ++j) {
      int et = wave + 8 * j;                       // e-tile 0..31
      int ebase = et * 16;
      v8f gi = zero8(), gf = zero8(), gg = zero8(), go = zero8();
      for (int kb = 0; kb < KZ; kb += 32) {
        v16bf a  = load_frag(zsm, KZ, 0, kb, lane);                  // LDS
        v16bf bi = load_frag(Wz, KZ, 0 * E_ + ebase, kb, lane);
        v16bf bf = load_frag(Wz, KZ, 1 * E_ + ebase, kb, lane);
        v16bf bg = load_frag(Wz, KZ, 2 * E_ + ebase, kb, lane);
        v16bf bo = load_frag(Wz, KZ, 3 * E_ + ebase, kb, lane);
        gi = wmma_bf16(a, bi, gi);
        gf = wmma_bf16(a, bf, gf);
        gg = wmma_bf16(a, bg, gg);
        go = wmma_bf16(a, bo, go);
      }
      int n = ebase + (lane & 15);
      float bi_i = b_ih[n]            + b_hh[n];
      float bi_f = b_ih[E_ + n]       + b_hh[E_ + n];
      float bi_g = b_ih[2 * E_ + n]   + b_hh[2 * E_ + n];
      float bi_o = b_ih[3 * E_ + n]   + b_hh[3 * E_ + n];
      int m0 = (lane >> 4) << 3;
      v8f hq;
#pragma unroll
      for (int i = 0; i < 8; ++i) {
        float iv = sigmoidf_(gi[i] + bi_i);
        float fv = sigmoidf_(gf[i] + bi_f);
        float gv = tanhf(gg[i] + bi_g);
        float ov = sigmoidf_(go[i] + bi_o);
        float c  = fv * cst[j][i] + iv * gv;
        cst[j][i] = c;
        hq[i] = ov * tanhf(c);
      }
      hreg[j] = hq;
      if (t >= 1) {
        float* hdst = hs + (size_t)(t - 1) * B_ * E_;
#pragma unroll
        for (int i = 0; i < 8; ++i)
          hdst[(size_t)(mrow + m0 + i) * E_ + n] = hq[i];
      }
    }
    __syncthreads();                               // all zsm reads done
    for (int j = 0; j < 4; ++j) {                  // publish h for next step
      int n = (wave + 8 * j) * 16 + (lane & 15);
      int m0 = (lane >> 4) << 3;
#pragma unroll
      for (int i = 0; i < 8; ++i)
        zsm[(m0 + i) * KZ + E_ + n] = (__bf16)hreg[j][i];
    }
  }
}

// ---------------- kernel 6: batch-norm over B + dropout -> bf16 ----------------
__global__ void nic_norm(const float* __restrict__ hs,
                         const float* __restrict__ gamma,
                         const float* __restrict__ beta,
                         const int* __restrict__ drop_mask,
                         __bf16* __restrict__ hs_bf) {
  int idx = blockIdx.x * blockDim.x + threadIdx.x;
  if (idx >= S_ * E_) return;
  int s = idx / E_, e = idx % E_;
  const float* p = hs + (size_t)s * B_ * E_ + e;
  float mean = 0.f;
  for (int b = 0; b < B_; ++b) mean += p[(size_t)b * E_];
  mean *= (1.0f / B_);
  float var = 0.f;
  for (int b = 0; b < B_; ++b) { float d = p[(size_t)b * E_] - mean; var += d * d; }
  var *= (1.0f / B_);
  float rs = rsqrtf(var + 1e-5f);
  float ga = gamma[e], be = beta[e];
  const float keep_inv = 1.0f / 0.7f;
  for (int b = 0; b < B_; ++b) {
    float v = (p[(size_t)b * E_] - mean) * rs * ga + be;
    v *= drop_mask[(size_t)b * E_ + e] ? keep_inv : 0.f;
    hs_bf[((size_t)s * B_ + b) * E_ + e] = (__bf16)v;
  }
}

// ---------------- kernel 7: decoder GEMM (2944 x 32000 x 512, WMMA) ----------------
// One wave computes a 32x64 output block (2 M-tiles x 4 N-tiles):
// 8 WMMAs per 6 fragment loads; W_dec read once per 32 output rows.
// preds is streamed with non-temporal stores so it does not evict W_dec from L2.
__global__ void __launch_bounds__(256)
nic_decoder(const __bf16* __restrict__ A,      // hs_bf [s*128+b][512]
            const __bf16* __restrict__ Bm,     // W_dec_bf [32000][512]
            const float* __restrict__ b_dec,
            float* __restrict__ out) {
  int tid = threadIdx.x, wave = tid >> 5, lane = tid & 31;
  long wid = (long)blockIdx.x * 8 + wave;      // one wave = 32x64 output block
  int mt = (int)(wid / 500);                   // 92 M-blocks of 32 rows
  int nt = (int)(wid % 500);                   // 500 strips of 64 cols
  if (mt >= 92) return;
  int mbase = mt * 32, nbase = nt * 64;
  v8f acc[2][4];
#pragma unroll
  for (int v = 0; v < 2; ++v)
#pragma unroll
    for (int u = 0; u < 4; ++u) acc[v][u] = zero8();

  for (int kb = 0; kb < E_; kb += 32) {
    v16bf a0 = load_frag(A, E_, mbase,      kb, lane);
    v16bf a1 = load_frag(A, E_, mbase + 16, kb, lane);
    // prefetch next k-slab of the B stream (global_prefetch_b8)
    if (kb + 64 < E_)
      __builtin_prefetch(Bm + (size_t)(nbase + (lane & 15)) * E_ + kb + 64, 0, 1);
#pragma unroll
    for (int u = 0; u < 4; ++u) {
      v16bf b = load_frag(Bm, E_, nbase + u * 16, kb, lane);
      acc[0][u] = wmma_bf16(a0, b, acc[0][u]);
      acc[1][u] = wmma_bf16(a1, b, acc[1][u]);
    }
  }
  int m0 = (lane >> 4) << 3;
#pragma unroll
  for (int v = 0; v < 2; ++v) {
#pragma unroll
    for (int u = 0; u < 4; ++u) {
      int n = nbase + u * 16 + (lane & 15);
      float bias = b_dec[n];
#pragma unroll
      for (int i = 0; i < 8; ++i) {
        int arow = mbase + v * 16 + m0 + i;    // = s*128 + b
        int s = arow >> 7, b = arow & 127;
        __builtin_nontemporal_store(acc[v][u][i] + bias,
                                    out + ((size_t)(b * S_ + s)) * V_ + n);
      }
    }
  }
}

// ---------------- launcher ----------------
extern "C" void kernel_launch(void* const* d_in, const int* in_sizes, int n_in,
                              void* d_out, int out_size, void* d_ws, size_t ws_size,
                              hipStream_t stream) {
  const float* img      = (const float*)d_in[0];
  const int*   cap      = (const int*)  d_in[1];
  const int*   cl       = (const int*)  d_in[2];
  const int*   dmask    = (const int*)  d_in[3];
  const float* W_img    = (const float*)d_in[4];
  const float* b_img    = (const float*)d_in[5];
  const float* emb      = (const float*)d_in[6];
  const float* W_ih     = (const float*)d_in[7];
  const float* b_ih     = (const float*)d_in[8];
  const float* W_hh     = (const float*)d_in[9];
  const float* b_hh     = (const float*)d_in[10];
  const float* gamma    = (const float*)d_in[11];
  const float* beta     = (const float*)d_in[12];
  const float* W_dec    = (const float*)d_in[13];
  const float* b_dec    = (const float*)d_in[14];

  char* ws = (char*)d_ws;
  int*    sort_ind = (int*)   (ws + OFF_SORT);
  __bf16* img_bf   = (__bf16*)(ws + OFF_IMGBF);
  __bf16* Wimg_bf  = (__bf16*)(ws + OFF_WIMG);
  __bf16* Wz_bf    = (__bf16*)(ws + OFF_WZ);
  __bf16* x_bf     = (__bf16*)(ws + OFF_XBF);
  float*  hs       = (float*) (ws + OFF_HS);
  __bf16* hs_bf    = (__bf16*)(ws + OFF_HSBF);
  __bf16* Wdec_bf  = (__bf16*)(ws + OFF_WDEC);

  float* out  = (float*)d_out;
  int*   outi = (int*)d_out;

  nic_prep<<<1, 128, 0, stream>>>(cl, cap, sort_ind, outi);
  nic_convert<<<4096, 256, 0, stream>>>(W_img, W_ih, W_hh, W_dec,
                                        Wimg_bf, Wz_bf, Wdec_bf);
  nic_gather<<<6912, 256, 0, stream>>>(img, cap, emb, sort_ind, img_bf, x_bf);
  nic_x0_gemm<<<32, 256, 0, stream>>>(img_bf, Wimg_bf, b_img, x_bf /* step 0 */);
  nic_lstm<<<8, 256, 0, stream>>>(x_bf, Wz_bf, b_ih, b_hh, hs);
  nic_norm<<<(S_ * E_ + 255) / 256, 256, 0, stream>>>(hs, gamma, beta, dmask, hs_bf);
  nic_decoder<<<5750, 256, 0, stream>>>(hs_bf, Wdec_bf, b_dec, out);
}
